// GraphAE_84473416777939
// MI455X (gfx1250) — matrix operand order repaired and checked by reading, not compiled, not run on hardware
//
#include <hip/hip_runtime.h>
#include <hip/hip_bf16.h>

#define N_NODES 50000
#define N_EDGES 800000
#define N_ETOT  (N_EDGES + N_NODES)   // with self-loops
#define FDIM    256
#define HEADS   4
#define LAT     64
#define NEG_SLOPE 0.2f

typedef __attribute__((ext_vector_type(16))) __bf16       v16bf;
typedef __attribute__((ext_vector_type(8)))  float        v8f;
typedef __attribute__((ext_vector_type(8)))  unsigned int v8u;

// ---------------------------------------------------------------------------
// Convert fp32 weight matrix (256x256, row-major [k][c]) to bf16, transposed
// to column-major [c][k] so LDS tiles give contiguous B fragments.
// ---------------------------------------------------------------------------
__global__ void k_cvt_w_T(const float* __restrict__ W, unsigned short* __restrict__ WbfT,
                          int n) {
    int i = blockIdx.x * blockDim.x + threadIdx.x;
    if (i >= n) return;
    int c = i >> 8;          // output column
    int k = i & 255;         // reduction index
    __bf16 v = (__bf16)W[k * FDIM + c];
    unsigned short u;
    __builtin_memcpy(&u, &v, 2);
    WbfT[(size_t)c * FDIM + k] = u;
}

// ---------------------------------------------------------------------------
// H = X @ W  (50000x256 @ 256x256), bf16 WMMA, fp32 accumulate.
// Block: 256 threads = 8 waves; wave w -> rows [128*bx + 16w, +16), cols
// [64*by, +64) as 4 accumulators. K swept in steps of 32.
// ---------------------------------------------------------------------------
__global__ __launch_bounds__(256)
void k_gemm(const float* __restrict__ X, const unsigned short* __restrict__ WbfT,
            float* __restrict__ H) {
    // weight tile: 64 cols x 256 K, bf16, column-major -> 32 KB LDS
    __shared__ __align__(32) unsigned short lds[64 * 256];

    const int tid = threadIdx.x;
    const int c0  = blockIdx.y * 64;
    const int r0b = blockIdx.x * 128;

    // cooperative load: thread t copies 128 B (64 bf16) of one column slice
    {
        const int col  = tid >> 2;
        const int part = tid & 3;
        const uint4* src = (const uint4*)(WbfT + (size_t)(c0 + col) * FDIM + part * 64);
        uint4*       dst = (uint4*)(lds + col * FDIM + part * 64);
        #pragma unroll
        for (int i = 0; i < 8; ++i) dst[i] = src[i];
    }
    __syncthreads();

    const int wave = tid >> 5;
    const int lane = tid & 31;
    const int half = lane >> 4;     // lane group 0: lanes 0-15, 1: lanes 16-31
    const int ml   = lane & 15;
    const int row  = r0b + wave * 16 + ml;   // A row for this lane

    v8f acc[4] = {};

    #pragma unroll
    for (int kb = 0; kb < FDIM; kb += 32) {
        // ---- A fragment: 16x32 bf16 in contiguous k order (same as B)
        // lanes 0-15 : elems 0..15 -> K=kb+0..15
        // lanes 16-31: elems 0..15 -> K=kb+16..31
        v16bf a;
        if (row < N_NODES) {
            const float* xr = X + (size_t)row * FDIM + kb + 16 * half;
            #pragma unroll
            for (int e = 0; e < 8; ++e) a[e]     = (__bf16)xr[e];
            #pragma unroll
            for (int e = 0; e < 8; ++e) a[8 + e] = (__bf16)xr[8 + e];
        } else {
            #pragma unroll
            for (int e = 0; e < 16; ++e) a[e] = (__bf16)0.0f;
        }

        // ---- 4 N-subtiles reuse the A fragment
        #pragma unroll
        for (int t = 0; t < 4; ++t) {
            // B fragment: 32x16 bf16; lane-half selects K base, col = ml.
            // Column-major LDS => 16 consecutive bf16 = one 32 B vector load.
            const v8u raw =
                *(const v8u*)(lds + (t * 16 + ml) * FDIM + kb + 16 * half);
            v16bf b = __builtin_bit_cast(v16bf, raw);
            acc[t] = __builtin_amdgcn_wmma_f32_16x16x32_bf16(
                false, a, false, b, (short)0, acc[t], false, false);
        }
    }

    // ---- store: C/D layout  M = v + 8*half, N = ml
    #pragma unroll
    for (int v = 0; v < 8; ++v) {
        const int rr = r0b + wave * 16 + half * 8 + v;
        if (rr < N_NODES) {
            #pragma unroll
            for (int t = 0; t < 4; ++t)
                H[(size_t)rr * FDIM + c0 + t * 16 + ml] = acc[t][v];
        }
    }
}

// ---------------------------------------------------------------------------
// Per-node attention logits: al_src[n,h] = <h[n,h,:], a_src[h,:]>, same dst.
// ---------------------------------------------------------------------------
__global__ void k_att(const float* __restrict__ H, const float* __restrict__ asv,
                      const float* __restrict__ adv, float* __restrict__ als,
                      float* __restrict__ ald) {
    int i = blockIdx.x * blockDim.x + threadIdx.x;
    if (i >= N_NODES * HEADS) return;
    int node = i >> 2, h = i & 3;
    const float* hr = H   + (size_t)node * FDIM + h * 64;
    const float* as = asv + h * 64;
    const float* ad = adv + h * 64;
    float s = 0.f, d = 0.f;
    #pragma unroll 8
    for (int o = 0; o < 64; ++o) { float v = hr[o]; s += v * as[o]; d += v * ad[o]; }
    als[i] = s;
    ald[i] = d;
}

__global__ void k_fill(float* __restrict__ p, float v, int n) {
    int i = blockIdx.x * blockDim.x + threadIdx.x;
    if (i < n) p[i] = v;
}

__device__ __forceinline__ void edge_sd(const int* __restrict__ ei, int e, int& s, int& d) {
    if (e < N_EDGES) { s = ei[e]; d = ei[N_EDGES + e]; }
    else             { s = d = e - N_EDGES; }          // self-loop
}
__device__ __forceinline__ float leaky(float x) { return x > 0.f ? x : NEG_SLOPE * x; }
// order-preserving float atomic max (init must be -inf bits)
__device__ __forceinline__ void atomicMaxF(float* addr, float val) {
    if (val >= 0.f) atomicMax((int*)addr,          __float_as_int(val));
    else            atomicMin((unsigned int*)addr, __float_as_uint(val));
}

__global__ void k_edge_max(const int* __restrict__ ei, const float* __restrict__ als,
                           const float* __restrict__ ald, float* __restrict__ m) {
    int e = blockIdx.x * blockDim.x + threadIdx.x;
    if (e >= N_ETOT) return;
    int s, d; edge_sd(ei, e, s, d);
    #pragma unroll
    for (int h = 0; h < HEADS; ++h) {
        float v = leaky(als[s * HEADS + h] + ald[d * HEADS + h]);
        atomicMaxF(&m[d * HEADS + h], v);
    }
}

__global__ void k_edge_exp(const int* __restrict__ ei, const float* __restrict__ als,
                           const float* __restrict__ ald, const float* __restrict__ m,
                           float* __restrict__ p, float* __restrict__ denom) {
    int e = blockIdx.x * blockDim.x + threadIdx.x;
    if (e >= N_ETOT) return;
    int s, d; edge_sd(ei, e, s, d);
    #pragma unroll
    for (int h = 0; h < HEADS; ++h) {
        float v  = leaky(als[s * HEADS + h] + ald[d * HEADS + h]);
        float pe = __expf(v - m[d * HEADS + h]);
        p[(size_t)e * HEADS + h] = pe;
        atomicAdd(&denom[d * HEADS + h], pe);
    }
}

// one thread per (edge, 4-feature chunk); 64 threads/edge cover 256 feats
__global__ void k_edge_scatter(const int* __restrict__ ei, const float* __restrict__ p,
                               const float* __restrict__ denom, const float* __restrict__ H,
                               float* __restrict__ agg) {
    long long tid = (long long)blockIdx.x * blockDim.x + threadIdx.x;
    if (tid >= (long long)N_ETOT * 64) return;
    int e = (int)(tid >> 6);
    int f = ((int)(tid & 63)) * 4;
    int h = f >> 6;
    int s, d; edge_sd(ei, e, s, d);
    float alpha = p[(size_t)e * HEADS + h] / denom[d * HEADS + h];
    const float4 hv = *(const float4*)(H + (size_t)s * FDIM + f);
    float* ag = agg + (size_t)d * FDIM + f;
    atomicAdd(ag + 0, alpha * hv.x);
    atomicAdd(ag + 1, alpha * hv.y);
    atomicAdd(ag + 2, alpha * hv.z);
    atomicAdd(ag + 3, alpha * hv.w);
}

__global__ void k_bias_elu(float* __restrict__ a, const float* __restrict__ b) {
    int i = blockIdx.x * blockDim.x + threadIdx.x;
    if (i >= N_NODES * FDIM) return;
    float v = a[i] + b[i & 255];
    a[i] = v > 0.f ? v : expm1f(v);
}

__global__ void k_mean_bias(const float* __restrict__ agg, const float* __restrict__ b2,
                            float* __restrict__ z) {
    int i = blockIdx.x * blockDim.x + threadIdx.x;
    if (i >= N_NODES * LAT) return;
    int node = i >> 6, o = i & 63;
    const float* ar = agg + (size_t)node * FDIM + o;
    z[i] = 0.25f * (ar[0] + ar[64] + ar[128] + ar[192]) + b2[o];
}

// one wave32 per edge; float2 per lane; shuffle-xor tree reduction
__global__ void k_decoder(const int* __restrict__ ei, const float* __restrict__ z,
                          float* __restrict__ out) {
    long long tid = (long long)blockIdx.x * blockDim.x + threadIdx.x;
    int e    = (int)(tid >> 5);
    int lane = (int)(tid & 31);
    if (e >= N_EDGES) return;
    int s = ei[e], d = ei[N_EDGES + e];
    const float2 a = *(const float2*)(z + (size_t)s * LAT + lane * 2);
    const float2 b = *(const float2*)(z + (size_t)d * LAT + lane * 2);
    float sum = a.x * b.x + a.y * b.y;
    #pragma unroll
    for (int off = 16; off > 0; off >>= 1) sum += __shfl_xor(sum, off, 32);
    if (lane == 0) out[e] = sum;
}

// ---------------------------------------------------------------------------
extern "C" void kernel_launch(void* const* d_in, const int* in_sizes, int n_in,
                              void* d_out, int out_size, void* d_ws, size_t ws_size,
                              hipStream_t stream) {
    const float* x     = (const float*)d_in[0];
    const int*   ei    = (const int*)  d_in[1];
    const float* W1    = (const float*)d_in[2];
    const float* asrc1 = (const float*)d_in[3];
    const float* adst1 = (const float*)d_in[4];
    const float* b1    = (const float*)d_in[5];
    const float* W2    = (const float*)d_in[6];
    const float* asrc2 = (const float*)d_in[7];
    const float* adst2 = (const float*)d_in[8];
    const float* b2    = (const float*)d_in[9];
    float* out = (float*)d_out;
    (void)in_sizes; (void)n_in; (void)out_size; (void)ws_size;

    char* ws = (char*)d_ws;
    size_t off = 0;
    auto carve = [&](size_t bytes) -> char* {
        char* p = ws + off;
        off += (bytes + 255) & ~(size_t)255;
        return p;
    };
    unsigned short* wbf1 = (unsigned short*)carve((size_t)FDIM * FDIM * 2);
    unsigned short* wbf2 = (unsigned short*)carve((size_t)FDIM * FDIM * 2);
    float* hbuf = (float*)carve((size_t)N_NODES * FDIM * 4);   // h1, reused as h2
    float* agg  = (float*)carve((size_t)N_NODES * FDIM * 4);   // agg1 -> hmid -> agg2
    float* zbuf = (float*)carve((size_t)N_NODES * LAT  * 4);
    float* als  = (float*)carve((size_t)N_NODES * HEADS * 4);
    float* ald  = (float*)carve((size_t)N_NODES * HEADS * 4);
    float* mbuf = (float*)carve((size_t)N_NODES * HEADS * 4);
    float* dbuf = (float*)carve((size_t)N_NODES * HEADS * 4);
    float* pbuf = (float*)carve((size_t)N_ETOT  * HEADS * 4);

    const int T = 256;
    const float NEG_INF = -__builtin_huge_valf();
    dim3 gemmGrid((N_NODES + 127) / 128, FDIM / 64);
    const int nhBlk  = (N_NODES * HEADS + T - 1) / T;
    const int nfBlk  = (N_NODES * FDIM  + T - 1) / T;
    const int etBlk  = (N_ETOT + T - 1) / T;
    const unsigned scBlk = (unsigned)(((long long)N_ETOT * 64 + T - 1) / T);

    // weights -> bf16 (transposed)
    k_cvt_w_T<<<(FDIM * FDIM + T - 1) / T, T, 0, stream>>>(W1, wbf1, FDIM * FDIM);
    k_cvt_w_T<<<(FDIM * FDIM + T - 1) / T, T, 0, stream>>>(W2, wbf2, FDIM * FDIM);

    // ---- layer 1 -----------------------------------------------------------
    k_gemm<<<gemmGrid, T, 0, stream>>>(x, wbf1, hbuf);
    k_att<<<nhBlk, T, 0, stream>>>(hbuf, asrc1, adst1, als, ald);
    k_fill<<<nhBlk, T, 0, stream>>>(mbuf, NEG_INF, N_NODES * HEADS);
    k_fill<<<nhBlk, T, 0, stream>>>(dbuf, 0.f,     N_NODES * HEADS);
    k_fill<<<nfBlk, T, 0, stream>>>(agg,  0.f,     N_NODES * FDIM);
    k_edge_max    <<<etBlk, T, 0, stream>>>(ei, als, ald, mbuf);
    k_edge_exp    <<<etBlk, T, 0, stream>>>(ei, als, ald, mbuf, pbuf, dbuf);
    k_edge_scatter<<<scBlk, T, 0, stream>>>(ei, pbuf, dbuf, hbuf, agg);
    k_bias_elu<<<nfBlk, T, 0, stream>>>(agg, b1);                // agg -> hmid

    // ---- layer 2 -----------------------------------------------------------
    k_gemm<<<gemmGrid, T, 0, stream>>>(agg, wbf2, hbuf);         // h2 into hbuf
    k_att<<<nhBlk, T, 0, stream>>>(hbuf, asrc2, adst2, als, ald);
    k_fill<<<nhBlk, T, 0, stream>>>(mbuf, NEG_INF, N_NODES * HEADS);
    k_fill<<<nhBlk, T, 0, stream>>>(dbuf, 0.f,     N_NODES * HEADS);
    k_fill<<<nfBlk, T, 0, stream>>>(agg,  0.f,     N_NODES * FDIM); // after gemm2
    k_edge_max    <<<etBlk, T, 0, stream>>>(ei, als, ald, mbuf);
    k_edge_exp    <<<etBlk, T, 0, stream>>>(ei, als, ald, mbuf, pbuf, dbuf);
    k_edge_scatter<<<scBlk, T, 0, stream>>>(ei, pbuf, dbuf, hbuf, agg);
    k_mean_bias<<<(N_NODES * LAT + T - 1) / T, T, 0, stream>>>(agg, b2, zbuf);

    // ---- decoder -----------------------------------------------------------
    const unsigned decBlk = (unsigned)(((long long)N_EDGES * 32 + T - 1) / T);
    k_decoder<<<decBlk, T, 0, stream>>>(ei, zbuf, out);
}
